// GCN_19404662243941
// MI455X (gfx1250) — compile-verified, matrix operand order checked by reading, not run on hardware
//
#include <hip/hip_runtime.h>
#include <hip/hip_bf16.h>

typedef __attribute__((ext_vector_type(16))) _Float16 v16h;
typedef __attribute__((ext_vector_type(8)))  _Float16 v8h;
typedef __attribute__((ext_vector_type(8)))  float    v8f;

constexpr int IN_F  = 256;
constexpr int HID   = 128;
constexpr int OUT_F = 64;

// ---------------- elementwise / prep kernels ----------------

__global__ void k_fill(float* __restrict__ p, float v, int n) {
  int i = blockIdx.x * blockDim.x + threadIdx.x;
  if (i < n) p[i] = v;
}

__global__ void k_cvt_f16(const float* __restrict__ in, _Float16* __restrict__ out, int n) {
  int i = blockIdx.x * blockDim.x + threadIdx.x;
  if (i < n) out[i] = (_Float16)in[i];
}

// Wt[n*K + k] = (f16) W[k*N + n]   (row-major [K][N] -> f16 [N][K])
__global__ void k_cvt_transpose_f16(const float* __restrict__ W, _Float16* __restrict__ Wt,
                                    int K, int N) {
  int i = blockIdx.x * blockDim.x + threadIdx.x;
  if (i < K * N) {
    int n = i / K;
    int k = i - n * K;
    Wt[i] = (_Float16)W[k * N + n];
  }
}

__global__ void k_degree(const int* __restrict__ dst, float* __restrict__ deg, int e) {
  int i = blockIdx.x * blockDim.x + threadIdx.x;
  if (i < e) atomicAdd(&deg[dst[i]], 1.0f);
}

__global__ void k_rsqrt(float* __restrict__ d, int n) {
  int i = blockIdx.x * blockDim.x + threadIdx.x;
  if (i < n) {
    float v = d[i];                     // deg >= 1 (self loop), but keep guard
    d[i] = (v > 0.0f) ? rsqrtf(v) : 0.0f;
  }
}

__global__ void k_norm(const int* __restrict__ src, const int* __restrict__ dst,
                       const float* __restrict__ dinv, float* __restrict__ nrm, int e) {
  int i = blockIdx.x * blockDim.x + threadIdx.x;
  if (i < e) nrm[i] = dinv[src[i]] * dinv[dst[i]];
}

// ---------------- WMMA GEMM ----------------
// C[M, NT*16] = A[M,K](f16) x Wt[NT*16, K]^T (f16), f32 accumulate.
// One wave computes a full 16-row strip: 16 x (NT*16), so each A fragment is
// loaded exactly once. W^T is staged in LDS (row-padded 16B -> conflict-free).

__device__ __forceinline__ v16h load_frag(const _Float16* p) {
  // elements 0..7  <- K = kbase + 0..7
  // elements 8..15 <- K = kbase + 16..23     (ISA 16-bit A/B fragment layout)
  v8h lo = *(const v8h*)p;
  v8h hi = *(const v8h*)(p + 16);
  v16h r;
#pragma unroll
  for (int i = 0; i < 8; ++i) { r[i] = lo[i]; r[i + 8] = hi[i]; }
  return r;
}

__device__ __forceinline__ v16h load_frag_lds(const _Float16* p) {
  v8h lo = *(const v8h*)p;
  v8h hi = *(const v8h*)(p + 16);
  v16h r;
#pragma unroll
  for (int i = 0; i < 8; ++i) { r[i] = lo[i]; r[i + 8] = hi[i]; }
  return r;
}

template <int K, int NT>
__global__ void k_gemm_wmma(const _Float16* __restrict__ A, const _Float16* __restrict__ Wt,
                            float* __restrict__ C, int mtiles) {
  constexpr int NCOLS   = NT * 16;
  constexpr int LDS_ROW = K + 8;        // +16B pad: row stride 132 banks = 4 mod 64
  __shared__ _Float16 wlds[NCOLS * LDS_ROW];

  // cooperative stage of Wt [NCOLS][K] into padded LDS image (16B chunks)
  constexpr int CHUNKS_PER_ROW = K / 8;
  constexpr int TOTAL_CHUNKS   = NCOLS * CHUNKS_PER_ROW;
  for (int c = threadIdx.x; c < TOTAL_CHUNKS; c += blockDim.x) {
    int row = c / CHUNKS_PER_ROW;
    int col = c - row * CHUNKS_PER_ROW;
    *(v8h*)&wlds[row * LDS_ROW + col * 8] = *(const v8h*)&Wt[(size_t)row * K + col * 8];
  }
  __syncthreads();

  const int lane = threadIdx.x & 31;
  const int wave = threadIdx.x >> 5;
  const int mt   = blockIdx.x * 4 + wave;   // wave-uniform guard below -> EXEC all-1s
  const int half = lane >> 4;               // 0: K-halves {0-7,16-23}; 1: {8-15,24-31}
  const int l15  = lane & 15;

  if (mt < mtiles) {
    v8f acc[NT];
#pragma unroll
    for (int t = 0; t < NT; ++t) acc[t] = v8f{};

    const _Float16* arow = A + (size_t)(mt * 16 + l15) * K + half * 8;
    const _Float16* brow = &wlds[l15 * LDS_ROW + half * 8];

#pragma unroll
    for (int k = 0; k < K; k += 32) {
      v16h a = load_frag(arow + k);
#pragma unroll
      for (int t = 0; t < NT; ++t) {
        v16h b = load_frag_lds(brow + t * 16 * LDS_ROW + k);
        acc[t] = __builtin_amdgcn_wmma_f32_16x16x32_f16(
            /*neg_a=*/false, a, /*neg_b=*/false, b,
            /*c_mod=*/(short)0, acc[t], /*reuse_a=*/false, /*reuse_b=*/false);
      }
    }

    // C/D layout: element r -> row = r + 8*half, col = lane&15
    float* cp = C + (size_t)(mt * 16 + half * 8) * NCOLS + l15;
#pragma unroll
    for (int t = 0; t < NT; ++t) {
#pragma unroll
      for (int r = 0; r < 8; ++r) cp[(size_t)r * NCOLS + t * 16] = acc[t][r];
    }
  }
}

// ---------------- edge aggregation: one wave per edge ----------------

template <int VEC>   // VEC = F/32 floats per lane (4 for F=128, 2 for F=64)
__global__ void k_scatter(const float* __restrict__ h, const int* __restrict__ src,
                          const int* __restrict__ dst, const float* __restrict__ nrm,
                          float* __restrict__ out, int e, int F) {
  int g    = blockIdx.x * blockDim.x + threadIdx.x;
  int edge = g >> 5;
  int lane = g & 31;
  if (edge >= e) return;
  int   s = src[edge];
  int   d = dst[edge];
  float w = nrm[edge];
  const float* hp = h   + (size_t)s * F + lane * VEC;
  float*       op = out + (size_t)d * F + lane * VEC;
  if constexpr (VEC == 4) {
    float4 v = *(const float4*)hp;
    atomicAdd(&op[0], v.x * w);
    atomicAdd(&op[1], v.y * w);
    atomicAdd(&op[2], v.z * w);
    atomicAdd(&op[3], v.w * w);
  } else {
    float2 v = *(const float2*)hp;
    atomicAdd(&op[0], v.x * w);
    atomicAdd(&op[1], v.y * w);
  }
}

// self-loop: agg += h * dinv[node]^2  (stream-serialized after scatter -> plain RMW safe)
__global__ void k_selfloop(const float* __restrict__ h, const float* __restrict__ dinv,
                           float* __restrict__ agg, int total, int fshift) {
  int i = blockIdx.x * blockDim.x + threadIdx.x;
  if (i < total) {
    float di = dinv[i >> fshift];
    agg[i] += h[i] * di * di;
  }
}

__global__ void k_relu_bias_f16(const float* __restrict__ agg, const float* __restrict__ b,
                                _Float16* __restrict__ out, int total, int fmask) {
  int i = blockIdx.x * blockDim.x + threadIdx.x;
  if (i < total) out[i] = (_Float16)fmaxf(agg[i] + b[i & fmask], 0.0f);
}

__global__ void k_bias_out(const float* __restrict__ agg, const float* __restrict__ b,
                           float* __restrict__ out, int total, int fmask) {
  int i = blockIdx.x * blockDim.x + threadIdx.x;
  if (i < total) out[i] = agg[i] + b[i & fmask];
}

// ---------------- host orchestration ----------------

static inline int nblk(long long n, int tb) { return (int)((n + tb - 1) / tb); }

extern "C" void kernel_launch(void* const* d_in, const int* in_sizes, int n_in,
                              void* d_out, int out_size, void* d_ws, size_t ws_size,
                              hipStream_t stream) {
  const float* x  = (const float*)d_in[0];
  const int*   ei = (const int*)d_in[1];
  const float* W1 = (const float*)d_in[2];
  const float* b1 = (const float*)d_in[3];
  const float* W2 = (const float*)d_in[4];
  const float* b2 = (const float*)d_in[5];

  const int N = in_sizes[0] / IN_F;     // 100000
  const int E = in_sizes[1] / 2;        // 1600000
  const int* src = ei;                  // edge_index[0]
  const int* dst = ei + E;              // edge_index[1]

  // ---- workspace arena (with aliased reuse) ----
  char* p = (char*)d_ws;
  auto take = [&](size_t bytes) -> char* {
    char* r = p;
    p += (bytes + 255) & ~(size_t)255;
    return r;
  };
  _Float16* x16  = (_Float16*)take((size_t)N * IN_F * 2);   // reused later as h1_f16
  float*    hx1  = (float*)   take((size_t)N * HID * 4);    // reused later as hx2
  float*    agg1 = (float*)   take((size_t)N * HID * 4);    // reused later as agg2
  float*    dinv = (float*)   take((size_t)N * 4);          // deg, then rsqrt in place
  float*    nrm  = (float*)   take((size_t)E * 4);
  _Float16* w1t  = (_Float16*)take((size_t)HID * IN_F * 2);
  _Float16* w2t  = (_Float16*)take((size_t)OUT_F * HID * 2);
  _Float16* h16  = x16;   // alias: x16 dead after GEMM1
  float*    hx2  = hx1;   // alias: hx1 dead after selfloop1
  float*    agg2 = agg1;  // alias: agg1 dead after epilogue1

  const int TB = 256;
  const int mtiles = (N + 15) / 16;     // 6250 (exact)

  // normalization
  k_fill<<<nblk(N, TB), TB, 0, stream>>>(dinv, 1.0f, N);                 // self-loop weight
  k_degree<<<nblk(E, TB), TB, 0, stream>>>(dst, dinv, E);
  k_rsqrt<<<nblk(N, TB), TB, 0, stream>>>(dinv, N);
  k_norm<<<nblk(E, TB), TB, 0, stream>>>(src, dst, dinv, nrm, E);

  // precision conversion
  k_cvt_f16<<<nblk((long long)N * IN_F, TB), TB, 0, stream>>>(x, x16, N * IN_F);
  k_cvt_transpose_f16<<<nblk(IN_F * HID, TB), TB, 0, stream>>>(W1, w1t, IN_F, HID);
  k_cvt_transpose_f16<<<nblk(HID * OUT_F, TB), TB, 0, stream>>>(W2, w2t, HID, OUT_F);

  // ---- layer 1 ----
  k_gemm_wmma<IN_F, HID / 16><<<nblk(mtiles, 4), 128, 0, stream>>>(x16, w1t, hx1, mtiles);
  k_fill<<<nblk((long long)N * HID, TB), TB, 0, stream>>>(agg1, 0.0f, N * HID);
  k_scatter<4><<<nblk((long long)E * 32, TB), TB, 0, stream>>>(hx1, src, dst, nrm, agg1, E, HID);
  k_selfloop<<<nblk((long long)N * HID, TB), TB, 0, stream>>>(hx1, dinv, agg1, N * HID, 7);
  k_relu_bias_f16<<<nblk((long long)N * HID, TB), TB, 0, stream>>>(agg1, b1, h16, N * HID, HID - 1);

  // ---- layer 2 ----
  k_gemm_wmma<HID, OUT_F / 16><<<nblk(mtiles, 4), 128, 0, stream>>>(h16, w2t, hx2, mtiles);
  k_fill<<<nblk((long long)N * OUT_F, TB), TB, 0, stream>>>(agg2, 0.0f, N * OUT_F);
  k_scatter<2><<<nblk((long long)E * 32, TB), TB, 0, stream>>>(hx2, src, dst, nrm, agg2, E, OUT_F);
  k_selfloop<<<nblk((long long)N * OUT_F, TB), TB, 0, stream>>>(hx2, dinv, agg2, N * OUT_F, 6);
  k_bias_out<<<nblk((long long)N * OUT_F, TB), TB, 0, stream>>>(agg2, b2, (float*)d_out, N * OUT_F, OUT_F - 1);
}
